// cross_set_score_70935679861137
// MI455X (gfx1250) — matrix operand order repaired and checked by reading, not compiled, not run on hardware
//
#include <hip/hip_runtime.h>
#include <hip/hip_bf16.h>

// Problem constants (from reference)
#define SS   32          // S
#define NIT  128         // NITEM_MAX
#define DD   256         // D
#define NH   8           // heads
#define HS   64          // head size
#define EE   512         // NH*HS

typedef __attribute__((ext_vector_type(16))) __bf16 v16bf;
typedef __attribute__((ext_vector_type(8)))  __bf16 v8bf;
typedef __attribute__((ext_vector_type(4)))  __bf16 v4bf;
typedef __attribute__((ext_vector_type(8)))  float  v8f;
typedef __attribute__((ext_vector_type(4)))  float  v4f;

struct Pack32 { v8bf lo, hi; };

static __device__ __forceinline__ v16bf make16(v8bf lo, v8bf hi) {
    Pack32 p{lo, hi};
    return __builtin_bit_cast(v16bf, p);
}

static __device__ __forceinline__ v8f wmma_bf16(v16bf a, v16bf b, v8f c) {
    return __builtin_amdgcn_wmma_f32_16x16x32_bf16(
        /*neg_a=*/false, a, /*neg_b=*/false, b,
        /*c_mod=*/(short)0, c, /*reuse_a=*/false, /*reuse_b=*/false);
}

// ---------------------------------------------------------------------------
// Kernel 0: W [256,512] f32 -> Wt [512,256] bf16 (transposed so WMMA B lanes
// read contiguous K). One-shot, tiny.
// ---------------------------------------------------------------------------
__global__ void wt_kernel(const float* __restrict__ W, __bf16* __restrict__ wt) {
    int n = blockIdx.x;    // 0..511 (output column of W)
    int k = threadIdx.x;   // 0..255
    wt[(size_t)n * DD + k] = (__bf16)W[(size_t)k * EE + n];
}

// ---------------------------------------------------------------------------
// Kernel 1: projection  xp[j,i,n,:] = x[j,i,n,:] @ W   (bf16 WMMA, f32 accum)
// Block = one (j,i) pair half (64 rows).  8 waves as 2(M) x 4(N):
// wave tile 32 x 128 = 2x8 tiles of 16x16, K = 256 in 8 chunks of 32.
// Epilogue: accum -> LDS (bf16) -> coalesced b128 global stores, 2 passes of
// 32 rows so the output staging buffer can alias the dead A-staging buffer.
// ---------------------------------------------------------------------------
#define LDSA_STRIDE 264   // 256 + 8 halfs pad -> 528B row pitch, kills bank conflicts
#define LDSO_STRIDE 520   // 512 + 8 halfs pad for output staging

__global__ __launch_bounds__(256) void proj_kernel(const float*  __restrict__ x,
                                                   const __bf16* __restrict__ wt,
                                                   __bf16*       __restrict__ xp) {
    __shared__ __bf16 smem[64 * LDSA_STRIDE];   // 33,792 B; reused for output staging

    const int blk  = blockIdx.x;     // 0..2047
    const int ji   = blk >> 1;       // j*32 + i
    const int half = blk & 1;        // item-half: rows 0..63 or 64..127
    const int tid  = threadIdx.x;

    // ---- stage 64x256 f32 tile -> bf16 LDS (coalesced b128 loads) ----
    const size_t xbase = ((size_t)ji * NIT + (size_t)half * 64) * DD;
#pragma unroll
    for (int it = 0; it < 16; ++it) {
        int idx = it * 1024 + tid * 4;             // flat over 64*256
        int row = idx >> 8;
        int col = idx & 255;
        v4f f = *(const v4f*)(x + xbase + idx);
        v4bf o;
        o[0] = (__bf16)f[0]; o[1] = (__bf16)f[1];
        o[2] = (__bf16)f[2]; o[3] = (__bf16)f[3];
        *(v4bf*)(smem + row * LDSA_STRIDE + col) = o;
    }
    __syncthreads();

    const int lane = tid & 31;
    const int wv   = tid >> 5;          // 0..7
    const int mw   = wv >> 2;           // 0..1  (M wave)
    const int nw   = wv & 3;            // 0..3  (N wave)
    const int lLow = lane & 15;
    const int kA   = (lane & 16) ? 16 : 0;   // A-operand K half select
    const int kB   = (lane & 16) ? 16 : 0;   // B-operand K half select

    v8f zero = {};
    v8f acc[2][8];
#pragma unroll
    for (int mt = 0; mt < 2; ++mt)
#pragma unroll
        for (int nt = 0; nt < 8; ++nt) acc[mt][nt] = zero;

    for (int kb = 0; kb < DD; kb += 32) {
        v16bf afrag[2];
#pragma unroll
        for (int mt = 0; mt < 2; ++mt) {
            const __bf16* p = smem + (mw * 32 + mt * 16 + lLow) * LDSA_STRIDE + kb + kA;
            afrag[mt] = make16(*(const v8bf*)p, *(const v8bf*)(p + 8));
        }
        v16bf bfrag[8];
#pragma unroll
        for (int nt = 0; nt < 8; ++nt) {
            const __bf16* p = wt + (size_t)(nw * 128 + nt * 16 + lLow) * DD + kb + kB;
            bfrag[nt] = *(const v16bf*)p;   // 16 contiguous bf16 (32B)
        }
#pragma unroll
        for (int mt = 0; mt < 2; ++mt)
#pragma unroll
            for (int nt = 0; nt < 8; ++nt)
                acc[mt][nt] = wmma_bf16(afrag[mt], bfrag[nt], acc[mt][nt]);
    }

    // ---- epilogue: stage bf16 results in LDS, drain with coalesced b128 ----
    // C/D layout: VGPR v -> M = v + (lane>=16 ? 8 : 0), N = lane&15.
    const int rowHi = (lane & 16) ? 8 : 0;
    __syncthreads();                         // everyone done reading A tile
#pragma unroll
    for (int p = 0; p < 2; ++p) {            // 32-row pass (waves with mw==p own it)
        if (mw == p) {
#pragma unroll
            for (int mt = 0; mt < 2; ++mt) {
#pragma unroll
                for (int nt = 0; nt < 8; ++nt) {
                    int col = nw * 128 + nt * 16 + lLow;
#pragma unroll
                    for (int v = 0; v < 8; ++v) {
                        int rloc = mt * 16 + v + rowHi;        // 0..31
                        smem[rloc * LDSO_STRIDE + col] = (__bf16)acc[mt][nt][v];
                    }
                }
            }
        }
        __syncthreads();
        // cooperative coalesced copy: 32 rows x 512 cols = 2048 b128 chunks
#pragma unroll
        for (int c = 0; c < 8; ++c) {
            int chunk = c * 256 + tid;        // 0..2047
            int row   = chunk >> 6;           // 64 chunks per row
            int col8  = chunk & 63;
            v8bf d = *(const v8bf*)(smem + row * LDSO_STRIDE + col8 * 8);
            size_t grow = (size_t)ji * NIT + half * 64 + p * 32 + row;
            *(v8bf*)(xp + grow * EE + col8 * 8) = d;
        }
        __syncthreads();                      // before next pass overwrites staging
    }
}

// ---------------------------------------------------------------------------
// Kernel 2: per (j,i): for each head h (one wave each):
//   P = xp[j,i,h] @ xp[i,j,h]^T  (128x128, K=64), sum(relu(P))/8,
// then combine heads with w2 and divide by nItem[j]*nItem[i].
// B operand (xp[i,j,h]) kept entirely in registers (16 frags = 128 VGPRs).
// ---------------------------------------------------------------------------
__global__ __launch_bounds__(256) void attn_kernel(const __bf16* __restrict__ xp,
                                                   const float*  __restrict__ nItem,
                                                   const float*  __restrict__ w2,
                                                   float*        __restrict__ out) {
    __shared__ float headSum[NH];

    const int ji  = blockIdx.x;          // 0..1023
    const int j   = ji >> 5;
    const int i   = ji & 31;
    const int tid = threadIdx.x;
    const int lane = tid & 31;
    const int h    = tid >> 5;           // head == wave id
    const int lLow = lane & 15;
    const int kA   = (lane & 16) ? 16 : 0;
    const int kB   = (lane & 16) ? 16 : 0;

    const __bf16* A = xp + ((size_t)(j * SS + i) * NIT) * EE + h * HS;  // [128 rows, stride EE]
    const __bf16* B = xp + ((size_t)(i * SS + j) * NIT) * EE + h * HS;

    // Preload all B fragments: 8 n-tiles x 2 k-chunks (X @ Y^T: B lane = row of Y)
    v16bf bfrag[8][2];
#pragma unroll
    for (int nt = 0; nt < 8; ++nt)
#pragma unroll
        for (int kc = 0; kc < 2; ++kc)
            bfrag[nt][kc] = *(const v16bf*)(B + (size_t)(nt * 16 + lLow) * EE + kc * 32 + kB);

    float rsum = 0.f;
    for (int strip = 0; strip < 8; ++strip) {          // M strips of 16 rows
        v16bf afrag[2];
#pragma unroll
        for (int kc = 0; kc < 2; ++kc) {
            const __bf16* p = A + (size_t)(strip * 16 + lLow) * EE + kc * 32 + kA;
            afrag[kc] = make16(*(const v8bf*)p, *(const v8bf*)(p + 8));
        }
#pragma unroll
        for (int nt = 0; nt < 8; ++nt) {
            v8f acc = {};
            acc = wmma_bf16(afrag[0], bfrag[nt][0], acc);
            acc = wmma_bf16(afrag[1], bfrag[nt][1], acc);
#pragma unroll
            for (int v = 0; v < 8; ++v) rsum += fmaxf(acc[v], 0.f);
        }
    }

    // wave32 cross-lane reduction
#pragma unroll
    for (int off = 16; off > 0; off >>= 1)
        rsum += __shfl_xor(rsum, off, 32);
    if (lane == 0) headSum[h] = rsum;
    __syncthreads();

    if (tid == 0) {
        float t = 0.f;
#pragma unroll
        for (int hh = 0; hh < NH; ++hh) t += headSum[hh] * w2[hh];
        // relu(P/8) summed == relu(P) summed / 8  -> apply 1/sqrt(hs)=0.125 here
        out[ji] = t * 0.125f / (nItem[j] * nItem[i]);
    }
}

// ---------------------------------------------------------------------------
extern "C" void kernel_launch(void* const* d_in, const int* in_sizes, int n_in,
                              void* d_out, int out_size, void* d_ws, size_t ws_size,
                              hipStream_t stream) {
    (void)in_sizes; (void)n_in; (void)out_size; (void)ws_size;
    const float* x     = (const float*)d_in[0];
    const float* nItem = (const float*)d_in[1];
    const float* W     = (const float*)d_in[2];
    const float* w2    = (const float*)d_in[3];

    __bf16* wt = (__bf16*)d_ws;                                   // 512*256*2 = 256 KiB
    __bf16* xp = (__bf16*)((char*)d_ws + (size_t)EE * DD * 2);    // 32*32*128*512*2 = 128 MiB

    wt_kernel  <<<EE,        DD,  0, stream>>>(W, wt);
    proj_kernel<<<SS*SS*2,   256, 0, stream>>>(x, wt, xp);
    attn_kernel<<<SS*SS,     256, 0, stream>>>(xp, nItem, w2, (float*)d_out);
}
